// PortillaSimoncelliMasked_68865505624623
// MI455X (gfx1250) — matrix operand order repaired and checked by reading, not compiled, not run on hardware
//
#include <hip/hip_runtime.h>
#include <math.h>

// Portilla-Simoncelli masked texture statistics for MI455X (gfx1250).
// WMMA-based masked covariance / autocorrelation, TDM tile staging, LDS FFTs.

#define BB   2
#define KORI 4
#define NSH  41
#define PI_F 3.14159265358979323846f

typedef float    v8f  __attribute__((ext_vector_type(8)));
typedef float    v2f  __attribute__((ext_vector_type(2)));
typedef _Float16 v16h __attribute__((ext_vector_type(16)));
typedef unsigned int u32x4 __attribute__((ext_vector_type(4)));
typedef int      i32x4 __attribute__((ext_vector_type(4)));
typedef int      i32x8 __attribute__((ext_vector_type(8)));

#if defined(__AMDGCN__) && __has_builtin(__builtin_amdgcn_tensor_load_to_lds)
#define PSM_TDM 1
#else
#define PSM_TDM 0
#endif
#if __has_include(<hip/amd_detail/amd_gfx1250_TDM.h>)
#define PSM_TDM6 1
#else
#define PSM_TDM6 0
#endif
#if defined(__AMDGCN__) && __has_builtin(__builtin_amdgcn_wmma_f32_16x16x32_f16)
#define PSM_WMMA16 1
#else
#define PSM_WMMA16 0
#endif
#if defined(__AMDGCN__) && __has_builtin(__builtin_amdgcn_wmma_f32_16x16x4_f32)
#define PSM_WMMA4 1
#else
#define PSM_WMMA4 0
#endif

// ---------------------------------------------------------------- helpers
__device__ __forceinline__ float psm_clip(float x, float lo, float hi) {
    return fminf(fmaxf(x, lo), hi);
}
__device__ __forceinline__ float psm_lo(float lr) {
    return cosf(PI_F * 0.5f * (psm_clip(lr, -1.f, 0.f) + 1.f));
}
__device__ __forceinline__ float psm_hi(float lr) {
    return cosf(PI_F * 0.5f * psm_clip(lr, -1.f, 0.f));
}
__device__ __forceinline__ float freqc(int u, int nn) {
    int s = (u < nn / 2) ? u : u - nn;   // u==nn/2 -> -nn/2 -> coord -1
    return (float)s / (float)(nn / 2);
}
// shift table: tril_indices(9)-4, keep (sh!=sw)||(sh<0), append (0,0)
__device__ __forceinline__ void psm_shift(int s, int* oa, int* ob) {
    if (s >= NSH - 1) { *oa = 0; *ob = 0; return; }
    int cnt = 0;
    for (int i = 0; i < 9; i++)
        for (int j = 0; j <= i; j++) {
            int sh = i - 4, sw = j - 4;
            if (sh != sw || sh < 0) {
                if (cnt == s) { *oa = sh; *ob = sw; return; }
                cnt++;
            }
        }
    *oa = 0; *ob = 0;
}

#if PSM_TDM
// Build a D# per CDNA5 ISA and fire the TDM.
__device__ __forceinline__ void tdm_load_tile(const float* gptr, void* ldsPtr,
                                              int tileW, int tileH, int tileZ,
                                              long rowPitch, long zPitch) {
    unsigned long long ga = (unsigned long long)(const void*)gptr;
    unsigned ldsOff = (unsigned)(unsigned long long)ldsPtr; // low 32b of flat LDS addr
    u32x4 g0;
    g0.x = 1u;                                         // count=1
    g0.y = ldsOff;                                     // lds_addr
    g0.z = (unsigned)(ga & 0xFFFFFFFFull);             // global_addr lo
    g0.w = (unsigned)((ga >> 32) & 0x1FFFFFFull) | (2u << 30); // hi + type=2
    unsigned long long s0 = (unsigned long long)rowPitch;
    unsigned long long s1 = (unsigned long long)zPitch;
    i32x8 g1;
    g1[0] = (int)(2u << 16);                                          // data_size=4B
    g1[1] = (int)(((unsigned)tileW & 0xFFFFu) << 16);                 // tensor_dim0 lo
    g1[2] = (int)((((unsigned)tileW >> 16) & 0xFFFFu) |
                  (((unsigned)tileH & 0xFFFFu) << 16));               // dim0 hi | dim1 lo
    g1[3] = (int)((((unsigned)tileH >> 16) & 0xFFFFu) |
                  (((unsigned)tileW & 0xFFFFu) << 16));               // dim1 hi | tile_dim0
    g1[4] = (int)(((unsigned)tileH & 0xFFFFu) |
                  (((unsigned)(tileZ > 1 ? tileZ : 0)) << 16));       // tile_dim1 | tile_dim2
    g1[5] = (int)(s0 & 0xFFFFFFFFull);
    g1[6] = (int)(((s0 >> 32) & 0xFFFFull) | ((s1 & 0xFFFFull) << 16));
    g1[7] = (int)((s1 >> 16) & 0xFFFFFFFFull);
    i32x4 g2 = {0, 0, 0, 0};
    i32x4 g3 = {0, 0, 0, 0};
    if (tileZ > 1) g2.x = tileZ;                                      // tensor_dim2
#if PSM_TDM6
    i32x8 g4 = {0, 0, 0, 0, 0, 0, 0, 0};
    __builtin_amdgcn_tensor_load_to_lds(g0, g1, g2, g3, g4, 0);
#else
    __builtin_amdgcn_tensor_load_to_lds(g0, g1, g2, g3, 0);
#endif
}
#endif

// ---------------------------------------------------------------- elementwise
__global__ void r2c_kernel(const float* __restrict__ in, float2* __restrict__ out, long n) {
    for (long i = (long)blockIdx.x * blockDim.x + threadIdx.x; i < n;
         i += (long)gridDim.x * blockDim.x)
        out[i] = make_float2(in[i], 0.f);
}
__global__ void c2real_kernel(const float2* __restrict__ in, float* __restrict__ out, long n) {
    for (long i = (long)blockIdx.x * blockDim.x + threadIdx.x; i < n;
         i += (long)gridDim.x * blockDim.x)
        out[i] = in[i].x;
}
__global__ void c2mag_kernel(const float2* __restrict__ in, float* __restrict__ out, long n) {
    for (long i = (long)blockIdx.x * blockDim.x + threadIdx.x; i < n;
         i += (long)gridDim.x * blockDim.x) {
        float2 v = in[i];
        out[i] = sqrtf(v.x * v.x + v.y * v.y);
    }
}

// binom5 blur + stride-2 down (pad_lo=1), *4, clip>=0
__global__ void blur_down_kernel(const float* __restrict__ in, float* __restrict__ out,
                                 int h, int w) {
    int h2 = h / 2, w2 = w / 2;
    long n = (long)3 * h2 * w2;
    const float bw[5] = {1.f / 16, 4.f / 16, 6.f / 16, 4.f / 16, 1.f / 16};
    for (long i = (long)blockIdx.x * blockDim.x + threadIdx.x; i < n;
         i += (long)gridDim.x * blockDim.x) {
        int m = (int)(i / ((long)h2 * w2));
        int p = (int)(i % ((long)h2 * w2));
        int y = p / w2, x = p % w2;
        float acc = 0.f;
        for (int dy = 0; dy < 5; dy++) {
            int iy = 2 * y - 1 + dy;
            if (iy < 0 || iy >= h) continue;
            for (int dx = 0; dx < 5; dx++) {
                int ix = 2 * x - 1 + dx;
                if (ix < 0 || ix >= w) continue;
                acc += bw[dy] * bw[dx] * in[((long)m * h + iy) * w + ix];
            }
        }
        float v = 4.f * acc;
        out[i] = v < 0.f ? 0.f : v;
    }
}

// ---------------------------------------------------------------- FFT (Stockham, LDS)
__global__ void __launch_bounds__(128) fft1d_kernel(const float2* __restrict__ in,
                                                    float2* __restrict__ out, int n,
                                                    int elemStride, int lineStride,
                                                    int linesPerImg, long imgStride,
                                                    float sign, float scale) {
    __shared__ float2 buf0[256];
    __shared__ float2 buf1[256];
    int line = blockIdx.x;
    long base = (long)(line / linesPerImg) * imgStride + (long)(line % linesPerImg) * lineStride;
    int t = threadIdx.x;
    for (int i = t; i < n; i += blockDim.x) buf0[i] = in[base + (long)i * elemStride];
    __syncthreads();
    float2* X = buf0;
    float2* Y = buf1;
    int s = 1, nn = n;
    while (nn > 1) {
        int m = nn >> 1;
        int half = n >> 1;
        for (int k = t; k < half; k += blockDim.x) {
            int p = k / s, q = k % s;
            float sn, cs;
            __sincosf(sign * 2.0f * PI_F * (float)p / (float)nn, &sn, &cs);
            float2 a = X[q + s * p];
            float2 b = X[q + s * (p + m)];
            float2 amb = make_float2(a.x - b.x, a.y - b.y);
            Y[q + s * (2 * p)] = make_float2(a.x + b.x, a.y + b.y);
            Y[q + s * (2 * p + 1)] = make_float2(amb.x * cs - amb.y * sn, amb.x * sn + amb.y * cs);
        }
        __syncthreads();
        float2* tp = X; X = Y; Y = tp;
        nn >>= 1; s <<= 1;
    }
    for (int i = t; i < n; i += blockDim.x) {
        float2 v = X[i];
        out[base + (long)i * elemStride] = make_float2(v.x * scale, v.y * scale);
    }
}

// ---------------------------------------------------------------- pyramid filters
__global__ void init_split_kernel(const float2* __restrict__ dft, float2* __restrict__ hp,
                                  float2* __restrict__ lo, int h, int w) {
    long hw = (long)h * w, n = (long)BB * hw;
    for (long i = (long)blockIdx.x * blockDim.x + threadIdx.x; i < n;
         i += (long)gridDim.x * blockDim.x) {
        int p = (int)(i % hw);
        int u = p / w, v = p % w;
        float fy = freqc(u, h), fx = freqc(v, w);
        float r = sqrtf(fx * fx + fy * fy);
        if (u == 0 && v == 0) r = 2.0f / (float)w;
        float lr = log2f(r);
        float2 d = dft[i];
        float gh = psm_hi(lr), gl = psm_lo(lr);
        hp[i] = make_float2(d.x * gh, d.y * gh);
        lo[i] = make_float2(d.x * gl, d.y * gl);
    }
}

__global__ void band_filter_kernel(const float2* __restrict__ lod, float2* __restrict__ bd,
                                   float2* __restrict__ lon, int h, int w) {
    long hw = (long)h * w, n = (long)BB * hw;
    const float amp = 2.f * 0.8944271909999159f; // 2*sqrt(0.8)
    for (long i = (long)blockIdx.x * blockDim.x + threadIdx.x; i < n;
         i += (long)gridDim.x * blockDim.x) {
        long b = i / hw;
        int p = (int)(i % hw);
        int u = p / w, v = p % w;
        float fy = freqc(u, h), fx = freqc(v, w);
        float r = sqrtf(fx * fx + fy * fy);
        if (u == 0 && v == 0) r = 2.0f / (float)w;
        float lr = log2f(r);
        float ang = atan2f(fy, fx);
        float hm = psm_hi(lr + 1.f);
        float lo1 = psm_lo(lr + 1.f);
        float2 L = lod[i];
        lon[i] = make_float2(L.x * lo1, L.y * lo1);
#pragma unroll
        for (int k = 0; k < KORI; k++) {
            float dth = ang - PI_F * (float)k / (float)KORI + PI_F;
            dth = dth - 2.f * PI_F * floorf(dth / (2.f * PI_F));
            dth -= PI_F;
            float am = 0.f;
            if (fabsf(dth) < PI_F * 0.5f) {
                float c = cosf(dth);
                am = amp * c * c * c;
            }
            float g = hm * am; // *(+1j): (x+iy)*i = (-y, x)
            bd[(b * KORI + k) * hw + p] = make_float2(-L.y * g, L.x * g);
        }
    }
}

__global__ void crop_decimate_kernel(const float2* __restrict__ in, float2* __restrict__ out,
                                     int h, int w) {
    int h2 = h / 2, w2 = w / 2;
    long hw2 = (long)h2 * w2, n = (long)BB * hw2;
    for (long i = (long)blockIdx.x * blockDim.x + threadIdx.x; i < n;
         i += (long)gridDim.x * blockDim.x) {
        long b = i / hw2;
        int p = (int)(i % hw2);
        int u2 = p / w2, v2 = p % w2;
        int u = (((u2 + h2 / 2) % h2) + h / 4 + h / 2) % h;
        int v = (((v2 + w2 / 2) % w2) + w / 4 + w / 2) % w;
        float2 s = in[(b * h + u) * (long)w + v];
        out[i] = make_float2(s.x * 0.25f, s.y * 0.25f);
    }
}

// bilinear 2x upsample of complex coarse coeffs + |up| + phase-double feats
__global__ void upsample_pd_kernel(const float2* __restrict__ cf, float* __restrict__ ua,
                                   float* __restrict__ ft, int h, int w, int h2, int w2) {
    long hw = (long)h * w, n = (long)BB * KORI * hw;
    for (long i = (long)blockIdx.x * blockDim.x + threadIdx.x; i < n;
         i += (long)gridDim.x * blockDim.x) {
        long bk = i / hw;
        int p = (int)(i % hw);
        int y = p / w, x = p % w;
        int b = (int)(bk / KORI), k = (int)(bk % KORI);
        float sy = (y + 0.5f) * 0.5f - 0.5f;
        float sx = (x + 0.5f) * 0.5f - 0.5f;
        int y0 = (int)floorf(sy), x0 = (int)floorf(sx);
        float wy = sy - (float)y0, wx = sx - (float)x0;
        int y0c = y0 < 0 ? 0 : (y0 >= h2 ? h2 - 1 : y0);
        int y1 = y0 + 1;
        int y1c = y1 < 0 ? 0 : (y1 >= h2 ? h2 - 1 : y1);
        int x0c = x0 < 0 ? 0 : (x0 >= w2 ? w2 - 1 : x0);
        int x1 = x0 + 1;
        int x1c = x1 < 0 ? 0 : (x1 >= w2 ? w2 - 1 : x1);
        const float2* base = cf + ((long)b * KORI + k) * h2 * w2;
        float2 c00 = base[(long)y0c * w2 + x0c], c01 = base[(long)y0c * w2 + x1c];
        float2 c10 = base[(long)y1c * w2 + x0c], c11 = base[(long)y1c * w2 + x1c];
        float re = (1.f - wy) * ((1.f - wx) * c00.x + wx * c01.x) +
                   wy * ((1.f - wx) * c10.x + wx * c11.x);
        float im = (1.f - wy) * ((1.f - wx) * c00.y + wx * c01.y) +
                   wy * ((1.f - wx) * c10.y + wx * c11.y);
        float a = sqrtf(re * re + im * im);
        ua[i] = a;
        float dn = a + 1e-6f;
        ft[((long)b * 2 * KORI + k) * hw + p] = (re * re - im * im) / dn;
        ft[((long)b * 2 * KORI + KORI + k) * hw + p] = 2.f * re * im / dn;
    }
}

// ---------------------------------------------------------------- masked moments
__global__ void __launch_bounds__(256) masked_moments_kernel(const float* __restrict__ x,
                                                             long bStride,
                                                             const float* __restrict__ mask,
                                                             int h, int w,
                                                             float* __restrict__ raw) {
    int b = blockIdx.x;
    long n = (long)h * w;
    float acc[12];
#pragma unroll
    for (int i = 0; i < 12; i++) acc[i] = 0.f;
    for (long p = threadIdx.x; p < n; p += blockDim.x) {
        if (p + 2048 < n) __builtin_prefetch(&x[(long)b * bStride + p + 2048], 0, 0);
        float v = x[(long)b * bStride + p];
        float v2 = v * v, v3 = v2 * v, v4 = v2 * v2;
#pragma unroll
        for (int m = 0; m < 3; m++) {
            float mm = mask[(long)m * n + p];
            acc[m * 4 + 0] += mm * v;
            acc[m * 4 + 1] += mm * v2;
            acc[m * 4 + 2] += mm * v3;
            acc[m * 4 + 3] += mm * v4;
        }
    }
    __shared__ float red[256];
    for (int i = 0; i < 12; i++) {
        red[threadIdx.x] = acc[i];
        __syncthreads();
        for (int st = 128; st > 0; st >>= 1) {
            if ((int)threadIdx.x < st) red[threadIdx.x] += red[threadIdx.x + st];
            __syncthreads();
        }
        if (threadIdx.x == 0) raw[((long)b * 3 + (i >> 2)) * 4 + (i & 3)] = red[0];
        __syncthreads();
    }
}

__global__ void finalize_moments_kernel(const float* __restrict__ raw, float* __restrict__ out,
                                        int F, int off, float eps, int mode) {
    int i = blockIdx.x * blockDim.x + threadIdx.x;
    if (i >= BB * 3) return;
    int b = i / 3, m = i % 3;
    const float* r = raw + (long)(b * 3 + m) * 4;
    float mu = r[0], e2 = r[1], e3 = r[2], e4 = r[3];
    float m2 = e2 - mu * mu;
    float m3 = e3 - 3.f * mu * e2 + 2.f * mu * mu * mu;
    float m4 = e4 - 4.f * mu * e3 + 6.f * mu * mu * e2 - 3.f * mu * mu * mu * mu;
    float sk = m3 / powf(m2 + eps, 1.5f);
    float ku = m4 / ((m2 + eps) * (m2 + eps));
    float* ob = out + (long)b * F;
    if (mode == 2) {
        if (off + m < F) ob[off + m] = mu;
        if (off + 3 + m < F) ob[off + 3 + m] = m2;
        if (off + 6 + m < F) ob[off + 6 + m] = sk;
        if (off + 9 + m < F) ob[off + 9 + m] = ku;
    } else if (mode == 1) {
        if (off + m < F) ob[off + m] = m2;
        if (off + 3 + m < F) ob[off + 3 + m] = sk;
        if (off + 6 + m < F) ob[off + 6 + m] = ku;
    } else {
        if (off + m < F) ob[off + m] = m2;
    }
}

// ---------------------------------------------------------------- WMMA cross-covariance
// Per batch: 3 Gram matrices (one per mask) via V_WMMA_F32_16X16X32_F16.
// A rows: [n*mask * a_k (KA)] [n*mask (1)]  |  B cols: [b_l (KB)] [1 (1)]
// Branch-free fragment build: plane arrays padded to 16 zeroed LDS rows.
__global__ void __launch_bounds__(256) wmma_xcorr_kernel(
    const float* __restrict__ aBase, long aB, long aK, int aE, int KA,
    const float* __restrict__ bBase, long bB, long bK, int bE, int KB,
    const float* __restrict__ mask, int h, int w, float* __restrict__ Edump) {
    __shared__ float sPA[16 * 256];
    __shared__ float sPB[16 * 256];
    __shared__ float sMK[3 * 256];
    __shared__ float sD[768];
    int b = blockIdx.x;
    long n = (long)h * w;
    float msc = (float)n;
    int lane = threadIdx.x & 31;
    int wv = threadIdx.x >> 5, nw = blockDim.x >> 5;
    int r16 = lane & 15;
    float isA = (r16 == KA) ? 1.f : 0.f, notA = 1.f - isA;
    float isB = (r16 == KB) ? 1.f : 0.f, notB = 1.f - isB;
    v8f vz = {0.f, 0.f, 0.f, 0.f, 0.f, 0.f, 0.f, 0.f};
    v8f acc0 = vz, acc1 = vz, acc2 = vz;
    // zero padding rows once (rows >= KA / >= KB are never written afterwards)
    for (int i = threadIdx.x; i < 16 * 256; i += blockDim.x) { sPA[i] = 0.f; sPB[i] = 0.f; }
    __syncthreads();
    for (long t0 = 0; t0 < n; t0 += 256) {
        __syncthreads();
        for (int i = threadIdx.x; i < 256; i += blockDim.x) {
            long p = t0 + i;
            for (int k = 0; k < KA; k++)
                sPA[k * 256 + i] = aBase[(long)b * aB + (long)k * aK + p * aE];
            for (int k = 0; k < KB; k++)
                sPB[k * 256 + i] = bBase[(long)b * bB + (long)k * bK + p * bE];
            for (int m = 0; m < 3; m++) sMK[m * 256 + i] = mask[(long)m * n + p] * msc;
        }
        __syncthreads();
        int pb = wv * 32;                 // one 32-pixel K-chunk per wave
        int kb16 = (lane & 16) ? 16 : 0;  // B frag: K = kb16 + i
        int kb8 = (lane & 16) ? 8 : 0;    // A frag: K = kb16 + i
        v16h fB, fA0, fA1, fA2;
#pragma unroll
        for (int i = 0; i < 16; i++) {
            int kk = pb + kb16 + i;
            float bv = sPB[r16 * 256 + kk];
            fB[i] = (_Float16)(bv * notB + isB);
        }
#pragma unroll
        for (int i = 0; i < 16; i++) {
            int kk = pb + kb16 + i;
            float av = sPA[r16 * 256 + kk];
            float ave = av * notA + isA;
            fA0[i] = (_Float16)(sMK[0 * 256 + kk] * ave);
            fA1[i] = (_Float16)(sMK[1 * 256 + kk] * ave);
            fA2[i] = (_Float16)(sMK[2 * 256 + kk] * ave);
        }
#if PSM_WMMA16
        acc0 = __builtin_amdgcn_wmma_f32_16x16x32_f16(false, fA0, false, fB, (short)0, acc0,
                                                      false, false);
        acc1 = __builtin_amdgcn_wmma_f32_16x16x32_f16(false, fA1, false, fB, (short)0, acc1,
                                                      false, false);
        acc2 = __builtin_amdgcn_wmma_f32_16x16x32_f16(false, fA2, false, fB, (short)0, acc2,
                                                      false, false);
#else
        (void)fA0; (void)fA1; (void)fA2; (void)fB;
#endif
    }
    __syncthreads();
    for (int i = threadIdx.x; i < 768; i += blockDim.x) sD[i] = 0.f;
    __syncthreads();
    for (int vv = 0; vv < nw; vv++) {
        if (wv == vv) {
#pragma unroll
            for (int g = 0; g < 8; g++) {
                int row = (lane < 16) ? g : g + 8;
                sD[0 * 256 + row * 16 + r16] += acc0[g];
                sD[1 * 256 + row * 16 + r16] += acc1[g];
                sD[2 * 256 + row * 16 + r16] += acc2[g];
            }
        }
        __syncthreads();
    }
    for (int i = threadIdx.x; i < 768; i += blockDim.x) Edump[(long)b * 768 + i] = sD[i];
}

__global__ void finalize_xcorr_kernel(const float* __restrict__ E, float* __restrict__ out,
                                      int F, int off, int KA, int KB, float invn) {
    int tot = BB * 3 * KA * KB;
    int i = blockIdx.x * blockDim.x + threadIdx.x;
    if (i >= tot) return;
    int b = i / (3 * KA * KB);
    int m = (i / (KA * KB)) % 3;
    int k = (i / KB) % KA;
    int l = i % KB;
    const float* e = E + ((long)b * 3 + m) * 256;
    float cov = (e[k * 16 + l] - e[k * 16 + KB] * e[KA * 16 + l] * invn) * invn;
    int o = off + (m * KA + k) * KB + l;
    if (o < F) out[(long)b * F + o] = cov;
}

__global__ void finalize_magmeans_kernel(const float* __restrict__ E, float* __restrict__ out,
                                         int F, int off, int KA, int KBcol, float invn) {
    int tot = BB * 3 * KA;
    int i = blockIdx.x * blockDim.x + threadIdx.x;
    if (i >= tot) return;
    int b = i / (3 * KA);
    int rm = i % (3 * KA);
    int m = rm / KA, k = rm % KA;
    float v = E[((long)b * 3 + m) * 256 + k * 16 + KBcol] * invn;
    int o = off + m * KA + k;
    if (o < F) out[(long)b * F + o] = v;
}

// ---------------------------------------------------------------- WMMA autocorrelation
// rows: {mask_m*x (3), mask_m (3)}; cols: 41 rolled copies of x (3 tiles of 16).
// f32 16x16x4 WMMA; LDS tile (8 rows + wrapped 4-halo) loaded by the TDM.
// Branch-free: unconditional LDS loads + multiply selects (lane constants).
__global__ void __launch_bounds__(256) wmma_autocorr_kernel(
    const float* __restrict__ xBase, long bStride, long cStride, int C,
    const float* __restrict__ mask, int h, int w, float* __restrict__ Ddump) {
    const int TRM = 8;
    int TR = h < TRM ? h : TRM;
    __shared__ float sX[(TRM + 8) * 256];
    __shared__ float sM[3 * TRM * 256];
    __shared__ float sD[768];
    int bc = blockIdx.x;
    const float* x = xBase + (long)(bc / C) * bStride + (long)(bc % C) * cStride;
    int lane = threadIdx.x & 31;
    int wv = threadIdx.x >> 5, nw = blockDim.x >> 5;
    int ncol = lane & 15;
    int kb = (lane & 16) ? 2 : 0;
    // lane constants: A-row role and per-tile shifts
    float aSelX = (ncol < 3) ? 1.f : 0.f;                  // rows 0..2: mask*x
    float aSelM = (ncol >= 3 && ncol < 6) ? 1.f : 0.f;     // rows 3..5: mask
    int mi = (ncol < 3) ? ncol : ((ncol < 6) ? ncol - 3 : 0);
    int sh0, sw0, sh1, sw1, sh2, sw2;
    psm_shift(0 * 16 + ncol, &sh0, &sw0);
    psm_shift(1 * 16 + ncol, &sh1, &sw1);
    psm_shift(2 * 16 + ncol, &sh2, &sw2);
    float v0f = (0 * 16 + ncol < NSH) ? 1.f : 0.f;
    float v1f = (1 * 16 + ncol < NSH) ? 1.f : 0.f;
    float v2f_ = (2 * 16 + ncol < NSH) ? 1.f : 0.f;
    v8f vz = {0.f, 0.f, 0.f, 0.f, 0.f, 0.f, 0.f, 0.f};
    v8f acc0 = vz, acc1 = vz, acc2 = vz;
    int nt = h / TR;
    int wm1 = w - 1;
    for (int rb = 0; rb < nt; rb++) {
        int r0 = rb * TR;
        __syncthreads();
        // wrapped halo rows (4 above + 4 below), cooperative
        for (int i = threadIdx.x; i < 8 * w; i += blockDim.x) {
            int hr = i / w, cc = i % w;
            int gr = (hr < 4) ? (r0 - 4 + hr + h) % h : (r0 + TR + hr - 4) % h;
            int lr2 = (hr < 4) ? hr : (TR + hr);
            sX[lr2 * w + cc] = x[(long)gr * w + cc];
        }
#if PSM_TDM
        if (wv == 0) {
            tdm_load_tile(x + (long)r0 * w, &sX[4 * w], w, TR, 1, w, 0);
            tdm_load_tile(mask + (long)r0 * w, &sM[0], w, TR, 3, w, (long)h * w);
#if __has_builtin(__builtin_amdgcn_s_wait_tensorcnt)
            __builtin_amdgcn_s_wait_tensorcnt(0);
#endif
        }
#else
        for (int i = threadIdx.x; i < TR * w; i += blockDim.x) {
            int rr = i / w, cc = i % w;
            sX[(rr + 4) * w + cc] = x[(long)(r0 + rr) * w + cc];
            for (int m = 0; m < 3; m++)
                sM[(m * TR + rr) * w + cc] = mask[((long)m * h + r0 + rr) * w + cc];
        }
#endif
        __syncthreads();
        int npix = TR * w;
#pragma unroll 2
        for (int p0 = wv * 4; p0 < npix; p0 += nw * 4) {
            int pl = p0 + kb;
            v2f fA, fB0, fB1, fB2;
#pragma unroll
            for (int e = 0; e < 2; e++) {
                int p = pl + e;
                int pr = p / w;
                int pc = p & wm1;
                float xv = sX[(pr + 4) * w + pc];
                float mval = sM[(mi * TR + pr) * w + pc];
                fA[e] = mval * (xv * aSelX + aSelM);
                int rr0 = pr - sh0 + 4, cc0 = (pc - sw0 + w) & wm1;
                float t0 = sX[rr0 * w + cc0];
                int rr1 = pr - sh1 + 4, cc1 = (pc - sw1 + w) & wm1;
                float t1 = sX[rr1 * w + cc1];
                int rr2 = pr - sh2 + 4, cc2 = (pc - sw2 + w) & wm1;
                float t2 = sX[rr2 * w + cc2];
                fB0[e] = t0 * v0f;
                fB1[e] = t1 * v1f;
                fB2[e] = t2 * v2f_;
            }
#if PSM_WMMA4
            acc0 = __builtin_amdgcn_wmma_f32_16x16x4_f32(false, fA, false, fB0, (short)0, acc0,
                                                         false, false);
            acc1 = __builtin_amdgcn_wmma_f32_16x16x4_f32(false, fA, false, fB1, (short)0, acc1,
                                                         false, false);
            acc2 = __builtin_amdgcn_wmma_f32_16x16x4_f32(false, fA, false, fB2, (short)0, acc2,
                                                         false, false);
#else
            (void)fA; (void)fB0; (void)fB1; (void)fB2;
#endif
        }
    }
    __syncthreads();
    for (int i = threadIdx.x; i < 768; i += blockDim.x) sD[i] = 0.f;
    __syncthreads();
    for (int vv = 0; vv < nw; vv++) {
        if (wv == vv) {
#pragma unroll
            for (int g = 0; g < 8; g++) {
                int row = (lane < 16) ? g : g + 8;
                sD[row * 48 + 0 * 16 + ncol] += acc0[g];
                sD[row * 48 + 1 * 16 + ncol] += acc1[g];
                sD[row * 48 + 2 * 16 + ncol] += acc2[g];
            }
        }
        __syncthreads();
    }
    for (int i = threadIdx.x; i < 768; i += blockDim.x) Ddump[(long)bc * 768 + i] = sD[i];
}

__global__ void finalize_autocorr_kernel(const float* __restrict__ D, float* __restrict__ out,
                                         int F, int off, int C) {
    int tot = BB * C * 3 * NSH;
    int i = blockIdx.x * blockDim.x + threadIdx.x;
    if (i >= tot) return;
    int b = i / (C * 3 * NSH);
    int rem = i % (C * 3 * NSH);
    int c = rem / (3 * NSH);
    int r2 = rem % (3 * NSH);
    int m = r2 / NSH, s = r2 % NSH;
    const float* d = D + ((long)b * C + c) * 768;
    float v = d[m * 48 + s] - d[(3 + m) * 48 + 40] * d[(3 + m) * 48 + s];
    int o = off + (c * 3 + m) * NSH + s;
    if (o < F) out[(long)b * F + o] = v;
}

// ---------------------------------------------------------------- host driver
extern "C" void kernel_launch(void* const* d_in, const int* in_sizes, int n_in, void* d_out,
                              int out_size, void* d_ws, size_t ws_size, hipStream_t stream) {
    (void)in_sizes; (void)n_in;
    const float* img = (const float*)d_in[0];   // [2,1,256,256]
    const float* mask0 = (const float*)d_in[1]; // [3,256,256]
    float* out = (float*)d_out;
    hipMemsetAsync(d_out, 0, (size_t)out_size * sizeof(float), stream);
    int F = out_size / BB;
    if (F <= 0) return;

    // feature offsets (reference concat order)
    int off = 0;
    int offPix = off; off += 12;
    int offM1[4], offA[4], offX[4];
    for (int j = 0; j < 4; j++) { offM1[j] = off; off += 12; offA[j] = off; off += 492; offX[j] = off; off += 48; }
    int offRM[5], offRA[5];
    for (int j = 0; j < 5; j++) { offRM[j] = off; off += 9; offRA[j] = off; off += 123; }
    int offC1[3], offC2[3];
    for (int j = 0; j < 3; j++) { offC1[j] = off; off += 48; offC2[j] = off; off += 96; }
    int offHV = off; off += 3; // total 3315 per batch

    // workspace carve-up
    float* W = (float*)d_ws;
    size_t wo = 0;
    auto A = [&](size_t nel) { float* p = W + wo; wo += (nel + 1) & ~(size_t)1; return p; };
    const float* MK[5];
    MK[0] = mask0;
    float* mlev[5] = {nullptr, nullptr, nullptr, nullptr, nullptr};
    for (int j = 1; j < 5; j++) { int hj = 256 >> j; mlev[j] = A((size_t)3 * hj * hj); MK[j] = mlev[j]; }
    float* cImg = A((size_t)BB * 256 * 256 * 2);
    float* dftA = A((size_t)BB * 256 * 256 * 2);
    float* lodft = A((size_t)BB * 256 * 256 * 2);
    float* lodft2 = A((size_t)BB * 256 * 256 * 2);
    float* bdft = A((size_t)BB * KORI * 256 * 256 * 2);
    float* tmpC = A((size_t)BB * 256 * 256 * 2);
    float* coef[4];
    float* mg[4];
    for (int j = 0; j < 4; j++) {
        int hj = 256 >> j;
        coef[j] = A((size_t)BB * KORI * hj * hj * 2);
        mg[j] = A((size_t)BB * KORI * hj * hj);
    }
    float* rec[5];
    for (int j = 0; j < 5; j++) { int hj = 256 >> j; rec[j] = A((size_t)BB * hj * hj); }
    float* hpR = A((size_t)BB * 256 * 256);
    float* upabs = A((size_t)BB * KORI * 256 * 256);
    float* feats = A((size_t)BB * 2 * KORI * 256 * 256);
    float* Ebuf = A(1536);
    float* Dbuf = A((size_t)BB * KORI * 768);
    float* rawM = A(64);
    if (wo * sizeof(float) > ws_size) return; // scratch too small: deterministic no-op (zeros)

    auto G = [&](long nel) { return dim3((unsigned)((nel + 255) / 256)); };
    auto fft2f = [&](float* inF, float* outF, int h, int w, int nimg, int inv) {
        float sgn = inv ? 1.f : -1.f;
        fft1d_kernel<<<dim3((unsigned)(nimg * h)), 128, 0, stream>>>(
            (const float2*)inF, (float2*)outF, w, 1, w, h, (long)h * w, sgn,
            inv ? 1.f / (float)w : 1.f);
        fft1d_kernel<<<dim3((unsigned)(nimg * w)), 128, 0, stream>>>(
            (const float2*)outF, (float2*)outF, h, w, 1, w, (long)h * w, sgn,
            inv ? 1.f / (float)h : 1.f);
    };

    // mask pyramid
    for (int j = 0; j < 4; j++) {
        int hj = 256 >> j;
        blur_down_kernel<<<G((long)3 * (hj / 2) * (hj / 2)), 256, 0, stream>>>(MK[j], mlev[j + 1],
                                                                               hj, hj);
    }
    // forward transform + initial split
    r2c_kernel<<<G((long)BB * 65536), 256, 0, stream>>>(img, (float2*)cImg, (long)BB * 65536);
    fft2f(cImg, dftA, 256, 256, BB, 0);
    init_split_kernel<<<G((long)BB * 65536), 256, 0, stream>>>((const float2*)dftA, (float2*)bdft,
                                                               (float2*)lodft, 256, 256);
    fft2f(bdft, tmpC, 256, 256, BB, 1);
    c2real_kernel<<<G((long)BB * 65536), 256, 0, stream>>>((const float2*)tmpC, hpR,
                                                           (long)BB * 65536);
    // pyramid scales
    for (int j = 0; j <= 4; j++) {
        int h = 256 >> j;
        long nbh = (long)BB * h * h;
        fft2f(lodft, tmpC, h, h, BB, 1);
        c2real_kernel<<<G(nbh), 256, 0, stream>>>((const float2*)tmpC, rec[j], nbh);
        if (j < 4) {
            band_filter_kernel<<<G(nbh), 256, 0, stream>>>((const float2*)lodft, (float2*)bdft,
                                                           (float2*)lodft2, h, h);
            fft2f(bdft, coef[j], h, h, BB * KORI, 1);
            c2mag_kernel<<<G((long)BB * KORI * h * h), 256, 0, stream>>>(
                (const float2*)coef[j], mg[j], (long)BB * KORI * h * h);
            crop_decimate_kernel<<<G((long)BB * (h / 2) * (h / 2)), 256, 0, stream>>>(
                (const float2*)lodft2, (float2*)lodft, h, h);
        }
    }
    // pixel statistics
    masked_moments_kernel<<<BB, 256, 0, stream>>>(img, 65536L, MK[0], 256, 256, rawM);
    finalize_moments_kernel<<<1, 64, 0, stream>>>(rawM, out, F, offPix, 0.1f, 2);
    // per-scale magnitude statistics
    for (int j = 0; j < 4; j++) {
        int h = 256 >> j;
        long n = (long)h * h;
        float invn = 1.f / (float)n;
        wmma_xcorr_kernel<<<BB, 256, 0, stream>>>(mg[j], KORI * n, n, 1, 4, mg[j], KORI * n, n, 1,
                                                  4, MK[j], h, h, Ebuf);
        finalize_magmeans_kernel<<<1, 64, 0, stream>>>(Ebuf, out, F, offM1[j], 4, 4, invn);
        finalize_xcorr_kernel<<<1, 128, 0, stream>>>(Ebuf, out, F, offX[j], 4, 4, invn);
        wmma_autocorr_kernel<<<BB * KORI, 256, 0, stream>>>(mg[j], KORI * n, n, KORI, MK[j], h, h,
                                                            Dbuf);
        finalize_autocorr_kernel<<<G((long)BB * KORI * 3 * NSH), 256, 0, stream>>>(Dbuf, out, F,
                                                                                   offA[j], KORI);
    }
    // reconstructed lowpass statistics
    for (int j = 0; j <= 4; j++) {
        int h = 256 >> j;
        long n = (long)h * h;
        masked_moments_kernel<<<BB, 256, 0, stream>>>(rec[j], n, MK[j], h, h, rawM);
        finalize_moments_kernel<<<1, 64, 0, stream>>>(rawM, out, F, offRM[j], 1e-6f, 1);
        wmma_autocorr_kernel<<<BB, 256, 0, stream>>>(rec[j], n, 0, 1, MK[j], h, h, Dbuf);
        finalize_autocorr_kernel<<<1, 256, 0, stream>>>(Dbuf, out, F, offRA[j], 1);
    }
    // cross-scale correlations
    for (int j = 0; j < 3; j++) {
        int h = 256 >> j;
        long n = (long)h * h;
        float invn = 1.f / (float)n;
        upsample_pd_kernel<<<G((long)BB * KORI * n), 256, 0, stream>>>(
            (const float2*)coef[j + 1], upabs, feats, h, h, h / 2, h / 2);
        wmma_xcorr_kernel<<<BB, 256, 0, stream>>>(mg[j], KORI * n, n, 1, 4, upabs, KORI * n, n, 1,
                                                  4, MK[j], h, h, Ebuf);
        finalize_xcorr_kernel<<<1, 128, 0, stream>>>(Ebuf, out, F, offC1[j], 4, 4, invn);
        wmma_xcorr_kernel<<<BB, 256, 0, stream>>>(coef[j], KORI * n * 2, n * 2, 2, 4, feats,
                                                  8 * n, n, 1, 8, MK[j], h, h, Ebuf);
        finalize_xcorr_kernel<<<1, 256, 0, stream>>>(Ebuf, out, F, offC2[j], 4, 8, invn);
    }
    // highpass variance
    masked_moments_kernel<<<BB, 256, 0, stream>>>(hpR, 65536L, MK[0], 256, 256, rawM);
    finalize_moments_kernel<<<1, 64, 0, stream>>>(rawM, out, F, offHV, 0.f, 0);
}
